// BiGRU_69106023793007
// MI455X (gfx1250) — matrix operand order repaired and checked by reading, not compiled, not run on hardware
//
#include <hip/hip_runtime.h>
#include <hip/hip_bf16.h>
#include <math.h>

// ---------------------------------------------------------------------------
// BiGRU: B=128, T=512, D=H=512.  out = concat(h_fwd_final, h_bwd_final) [128,1024]
// Strategy: persistent kernel, 2 grid barriers per timestep.
//   phase 1 (per step): gi = x_t @ W_ih^T, gh = h @ W_hh^T  via v_wmma_f32_16x16x32_bf16
//   phase 2 (per step): gates + h update (fp32 elementwise)
// ---------------------------------------------------------------------------

typedef __bf16 v16bf __attribute__((ext_vector_type(16)));
typedef float  v8f   __attribute__((ext_vector_type(8)));

#define NBLK 48      // persistent blocks (must all be resident; trivial on MI455X)
#define TT   512
#define BB   128
#define DK   512     // D == H == 512 (shared K dim)
#define G3   1536    // 3*H
#define KT   16      // K tiles of 32  (512/32)
#define NTIL 96      // N tiles of 16  (1536/16)
#define WPACK (NTIL * KT * 512)   // packed bf16 elements per weight matrix

// ---- pack row-major fp32 weight [1536,512] into per-lane WMMA B layout -----
// B tile = 16(N) x 32(K). lane n<16: col n, K={k0..k0+7, k0+16..k0+23} ; lane n+16: col n, K={k0+8..k0+15, k0+24..k0+31}
// storage: tile (nt,kt) -> 512 contiguous bf16 at ((nt*KT + kt)*512 + lane*16 + j)
__global__ void pack_weights(const float* __restrict__ w0, const float* __restrict__ w1,
                             const float* __restrict__ w2, const float* __restrict__ w3,
                             __bf16* __restrict__ d0, __bf16* __restrict__ d1,
                             __bf16* __restrict__ d2, __bf16* __restrict__ d3) {
    const float* srcs[4] = {w0, w1, w2, w3};
    __bf16*      dsts[4] = {d0, d1, d2, d3};
    const float* w   = srcs[blockIdx.y];
    __bf16*      dst = dsts[blockIdx.y];

    int idx = blockIdx.x * 256 + threadIdx.x;      // [0, WPACK)
    if (idx >= WPACK) return;
    int j    = idx & 15;
    int lane = (idx >> 4) & 31;
    int kt   = (idx >> 9) & (KT - 1);
    int nt   = idx >> 13;
    int n = nt * 16 + (lane & 15);
    int k = kt * 32 + ((lane >> 4) ? 8 : 0) + (j & 7) + ((j >> 3) << 4);
    dst[idx] = (__bf16)w[(size_t)n * DK + k];
}

// ---- A fragment load: fp32 row -> bf16 per-lane A layout (16x32 bf16) ------
// lane l<16: row=l, K={k0..k0+7, k0+16..k0+23}; lane l+16: row=l, K={k0+8..15, k0+24..31}
// caller pre-offsets pointer by row*512 + (lhalf*8); here we read 8 floats at p
// and 8 floats at p+16.
__device__ __forceinline__ v16bf load_a16(const float* __restrict__ p) {
    float4 f0 = *(const float4*)(p);
    float4 f1 = *(const float4*)(p + 4);
    float4 f2 = *(const float4*)(p + 16);
    float4 f3 = *(const float4*)(p + 20);
    v16bf a;
    a[0]  = (__bf16)f0.x; a[1]  = (__bf16)f0.y; a[2]  = (__bf16)f0.z; a[3]  = (__bf16)f0.w;
    a[4]  = (__bf16)f1.x; a[5]  = (__bf16)f1.y; a[6]  = (__bf16)f1.z; a[7]  = (__bf16)f1.w;
    a[8]  = (__bf16)f2.x; a[9]  = (__bf16)f2.y; a[10] = (__bf16)f2.z; a[11] = (__bf16)f2.w;
    a[12] = (__bf16)f3.x; a[13] = (__bf16)f3.y; a[14] = (__bf16)f3.z; a[15] = (__bf16)f3.w;
    return a;
}

// ---- one wave computes a 16(M) x 64(N) tile of C = A(fp32,MxK) * Wpacked ---
__device__ __forceinline__ void wave_gemm(const float* __restrict__ Arow,   // row ptr + lhalf*8
                                          const __bf16* __restrict__ Wp,    // packed weight base
                                          float* __restrict__ Cbuf,         // [128 x 1536]
                                          int nb, int lane, int m_base, int n_base,
                                          int lmod, int lhalf) {
    v8f acc[4];
    const v8f vz = {0.f, 0.f, 0.f, 0.f, 0.f, 0.f, 0.f, 0.f};
#pragma unroll
    for (int i = 0; i < 4; ++i) acc[i] = vz;

    const __bf16* b0 = Wp + (size_t)(nb * 4 + 0) * (KT * 512) + lane * 16;
    const __bf16* b1 = b0 + (KT * 512);
    const __bf16* b2 = b1 + (KT * 512);
    const __bf16* b3 = b2 + (KT * 512);

#pragma unroll 4
    for (int kk = 0; kk < KT; ++kk) {
        v16bf a   = load_a16(Arow + kk * 32);
        v16bf vb0 = *(const v16bf*)(b0 + kk * 512);
        v16bf vb1 = *(const v16bf*)(b1 + kk * 512);
        v16bf vb2 = *(const v16bf*)(b2 + kk * 512);
        v16bf vb3 = *(const v16bf*)(b3 + kk * 512);
        acc[0] = __builtin_amdgcn_wmma_f32_16x16x32_bf16(false, a, false, vb0, (short)0, acc[0], false, false);
        acc[1] = __builtin_amdgcn_wmma_f32_16x16x32_bf16(false, a, false, vb1, (short)0, acc[1], false, false);
        acc[2] = __builtin_amdgcn_wmma_f32_16x16x32_bf16(false, a, false, vb2, (short)0, acc[2], false, false);
        acc[3] = __builtin_amdgcn_wmma_f32_16x16x32_bf16(false, a, false, vb3, (short)0, acc[3], false, false);
    }

    // C/D layout: VGPR r, lanes 0-15 -> row r, lanes 16-31 -> row r+8; col = lane%16
#pragma unroll
    for (int i = 0; i < 4; ++i) {
        float* cp = Cbuf + (size_t)(m_base + lhalf * 8) * G3 + (n_base + i * 16 + lmod);
#pragma unroll
        for (int r = 0; r < 8; ++r) cp[(size_t)r * G3] = acc[i][r];
    }
}

// ---- grid-wide sense-reversing barrier (all NBLK blocks resident) ----------
__device__ __forceinline__ void grid_barrier(int* cnt, int* gen) {
    __syncthreads();
    if (threadIdx.x == 0) {
        __threadfence();
        int g = __hip_atomic_load(gen, __ATOMIC_RELAXED, __HIP_MEMORY_SCOPE_AGENT);
        int a = __hip_atomic_fetch_add(cnt, 1, __ATOMIC_ACQ_REL, __HIP_MEMORY_SCOPE_AGENT);
        if (a == NBLK - 1) {
            __hip_atomic_store(cnt, 0, __ATOMIC_RELAXED, __HIP_MEMORY_SCOPE_AGENT);
            __hip_atomic_fetch_add(gen, 1, __ATOMIC_ACQ_REL, __HIP_MEMORY_SCOPE_AGENT);
        } else {
            while (__hip_atomic_load(gen, __ATOMIC_ACQUIRE, __HIP_MEMORY_SCOPE_AGENT) == g)
                __builtin_amdgcn_s_sleep(1);
        }
        __threadfence();
    }
    __syncthreads();
}

__global__ void __launch_bounds__(256)
bigru_persistent(const float* __restrict__ x,
                 const float* __restrict__ bihf, const float* __restrict__ bhhf,
                 const float* __restrict__ bihb, const float* __restrict__ bhhb,
                 const __bf16* __restrict__ pwihf, const __bf16* __restrict__ pwhhf,
                 const __bf16* __restrict__ pwihb, const __bf16* __restrict__ pwhhb,
                 float* __restrict__ hf, float* __restrict__ hb,
                 float* __restrict__ gif, float* __restrict__ ghf,
                 float* __restrict__ gib, float* __restrict__ ghb,
                 int* cnt, int* gen, float* __restrict__ out) {
    const int lane  = threadIdx.x & 31;
    const int lmod  = lane & 15;
    const int lhalf = lane >> 4;
    const int wid   = threadIdx.x >> 5;
    const int gw    = blockIdx.x * 8 + wid;     // 0..383 = 2 dirs * 8 mtiles * 24 nblocks
    const int dir   = gw / 192;
    const int rem   = gw % 192;
    const int mt    = rem / 24;
    const int nb    = rem % 24;
    const int m_base = mt * 16;
    const int n_base = nb * 64;

    const __bf16* Wih = dir ? pwihb : pwihf;
    const __bf16* Whh = dir ? pwhhb : pwhhf;
    float* Hm  = dir ? hb  : hf;
    float* GIm = dir ? gib : gif;
    float* GHm = dir ? ghb : ghf;

    // hoisted per-lane A row pointers (row = m_base + lmod for all lanes)
    const float* hrow = Hm + (size_t)(m_base + lmod) * DK + lhalf * 8;
    const int tid = blockIdx.x * 256 + threadIdx.x;

    for (int s = 0; s < TT; ++s) {
        const int t = dir ? (TT - 1 - s) : s;

        // --- phase 1: gi = x_t @ Wih^T ; gh = h @ Whh^T (WMMA bf16, fp32 acc)
        const float* xrow = x + ((size_t)(m_base + lmod) * TT + t) * DK + lhalf * 8;
        wave_gemm(xrow, Wih, GIm, nb, lane, m_base, n_base, lmod, lhalf);
        wave_gemm(hrow, Whh, GHm, nb, lane, m_base, n_base, lmod, lhalf);

        grid_barrier(cnt, gen);

        // --- phase 2: gates + hidden update (fp32 elementwise)
        for (int idx = tid; idx < 2 * BB * DK; idx += NBLK * 256) {
            int d1 = idx >> 16;               // BB*DK = 65536 per direction
            int rm = idx & 65535;
            int b  = rm >> 9;
            int j  = rm & 511;
            const float* GI  = d1 ? gib : gif;
            const float* GH  = d1 ? ghb : ghf;
            const float* BIH = d1 ? bihb : bihf;
            const float* BHH = d1 ? bhhb : bhhf;
            float*       He  = d1 ? hb : hf;
            size_t rb = (size_t)b * G3;
            float ir  = GI[rb + j]        + BIH[j];
            float hr  = GH[rb + j]        + BHH[j];
            float iz  = GI[rb + 512 + j]  + BIH[512 + j];
            float hz  = GH[rb + 512 + j]  + BHH[512 + j];
            float inn = GI[rb + 1024 + j] + BIH[1024 + j];
            float hn  = GH[rb + 1024 + j] + BHH[1024 + j];
            float r = 1.0f / (1.0f + __expf(-(ir + hr)));
            float z = 1.0f / (1.0f + __expf(-(iz + hz)));
            float n = tanhf(inn + r * hn);
            float hv  = He[(size_t)b * DK + j];
            float hnv = (1.0f - z) * n + z * hv;
            He[(size_t)b * DK + j] = hnv;
            if (s == TT - 1) out[(size_t)b * 1024 + d1 * 512 + j] = hnv;
        }

        grid_barrier(cnt, gen);
    }
}

extern "C" void kernel_launch(void* const* d_in, const int* in_sizes, int n_in,
                              void* d_out, int out_size, void* d_ws, size_t ws_size,
                              hipStream_t stream) {
    const float* x    = (const float*)d_in[0];
    const float* wihf = (const float*)d_in[1];
    const float* whhf = (const float*)d_in[2];
    const float* bihf = (const float*)d_in[3];
    const float* bhhf = (const float*)d_in[4];
    const float* wihb = (const float*)d_in[5];
    const float* whhb = (const float*)d_in[6];
    const float* bihb = (const float*)d_in[7];
    const float* bhhb = (const float*)d_in[8];

    // workspace layout (~10 MB)
    char* p = (char*)d_ws;
    int*    bar   = (int*)p;                        // [0]=count, [1]=generation
    __bf16* pwihf = (__bf16*)(p + 256);
    __bf16* pwhhf = pwihf + WPACK;
    __bf16* pwihb = pwhhf + WPACK;
    __bf16* pwhhb = pwihb + WPACK;
    float*  hf    = (float*)(pwhhb + WPACK);
    float*  hb    = hf + BB * DK;
    float*  gif   = hb + BB * DK;
    float*  ghf   = gif + BB * G3;
    float*  gib   = ghf + BB * G3;
    float*  ghb   = gib + BB * G3;

    // deterministic re-init each call (graph-capture safe)
    hipMemsetAsync(bar, 0, 256, stream);
    hipMemsetAsync(hf, 0, 2 * BB * DK * sizeof(float), stream);

    dim3 pg(WPACK / 256, 4);
    pack_weights<<<pg, 256, 0, stream>>>(wihf, whhf, wihb, whhb,
                                         pwihf, pwhhf, pwihb, pwhhb);

    bigru_persistent<<<NBLK, 256, 0, stream>>>(x, bihf, bhhf, bihb, bhhb,
                                               pwihf, pwhhf, pwihb, pwhhb,
                                               hf, hb, gif, ghf, gib, ghb,
                                               bar, bar + 1, (float*)d_out);
}
